// Encoder_81140522156752
// MI455X (gfx1250) — matrix operand order repaired and checked by reading, not compiled, not run on hardware
//
#include <hip/hip_runtime.h>

// ---------------------------------------------------------------------------
// Types for CDNA5 WMMA (wave32): V_WMMA_F32_16X16X32_BF16
// ---------------------------------------------------------------------------
typedef __bf16 bf16_t;
typedef __attribute__((ext_vector_type(16))) __bf16 v16bf;
typedef __attribute__((ext_vector_type(8)))  float  v8f;

#define NEG_SLOPE 0.2f

static __device__ __forceinline__ bf16_t f2bf(float f) {
  union { float f; unsigned u; } v; v.f = f;
  unsigned r = (v.u + 0x7FFFu + ((v.u >> 16) & 1u)) >> 16;   // RNE
  unsigned short h = (unsigned short)r;
  bf16_t out;
  __builtin_memcpy(&out, &h, sizeof(out));
  return out;
}

// float atomic-max via sign-aware integer atomics (works with -inf init)
static __device__ __forceinline__ void atomicMaxFloat(float* addr, float val) {
  if (val >= 0.0f) {
    atomicMax((int*)addr, __float_as_int(val));
  } else {
    atomicMin((unsigned int*)addr, __float_as_uint(val));
  }
}

// WMMA A-operand element address math (ISA 7.12.2, 16-bit A 16x32):
// lane 0-15 row M=lane, K 0-7 in v0-3 / 16-23 in v4-7; lanes 16-31 same rows,
// K halves 8-15 / 24-31. e = packed element index 0..15 (vgpr = e/2, half-word
// = e%2).
static __device__ __forceinline__ int a_kk(int e, int half) {
  const int vg = e >> 1, p = e & 1;
  return (vg < 4) ? (vg * 2 + p + half * 8) : (16 + (vg - 4) * 2 + p + half * 8);
}

// ---------------------------------------------------------------------------
// Utility kernels
// ---------------------------------------------------------------------------
__global__ void fill_f32_kernel(float* p, float v, long n) {
  long i = (long)blockIdx.x * blockDim.x + threadIdx.x;
  if (i < n) p[i] = v;
}

// final: out = acc + bias (H==1, mean over single head == identity)
__global__ void bias_out_kernel(const float* __restrict__ acc,
                                const float* __restrict__ bias,
                                float* __restrict__ out, long n, int F) {
  long i = (long)blockIdx.x * blockDim.x + threadIdx.x;
  if (i < n) out[i] = acc[i] + bias[i % F];
}

// ---------------------------------------------------------------------------
// Packing kernels: rearrange f32 matrices into the exact WMMA VGPR order so
// the GEMM inner loop is pure 128-bit contiguous loads.
//   packed layout: [tile][kblock][lane 0..31][e 0..15] bf16
// ---------------------------------------------------------------------------
// A: src row-major [M,K] f32, M % 16 == 0, K % 32 == 0
__global__ void pack_a_kernel(const float* __restrict__ src,
                              bf16_t* __restrict__ dst, int M, int K) {
  const int kblocks = K >> 5;
  long idx = (long)blockIdx.x * blockDim.x + threadIdx.x;
  if (idx >= (long)M * K) return;
  const int e    = (int)(idx & 15);
  const int lane = (int)((idx >> 4) & 31);
  const int kb   = (int)((idx >> 9) % kblocks);
  const int mt   = (int)(idx / ((long)kblocks << 9));
  const int row  = mt * 16 + (lane & 15);
  const int half = lane >> 4;
  const int kk   = (kb << 5) + a_kk(e, half);
  dst[idx] = f2bf(src[(size_t)row * K + kk]);
}

// A variant fused with relu(acc + bias) for the layer-2 input
__global__ void pack_a_relu_bias_kernel(const float* __restrict__ acc,
                                        const float* __restrict__ bias,
                                        bf16_t* __restrict__ dst, int M, int K) {
  const int kblocks = K >> 5;
  long idx = (long)blockIdx.x * blockDim.x + threadIdx.x;
  if (idx >= (long)M * K) return;
  const int e    = (int)(idx & 15);
  const int lane = (int)((idx >> 4) & 31);
  const int kb   = (int)((idx >> 9) % kblocks);
  const int mt   = (int)(idx / ((long)kblocks << 9));
  const int row  = mt * 16 + (lane & 15);
  const int half = lane >> 4;
  const int kk   = (kb << 5) + a_kk(e, half);
  float v = acc[(size_t)row * K + kk] + bias[kk];
  dst[idx] = f2bf(v > 0.0f ? v : 0.0f);
}

// B: src row-major [K,Nc] f32, Nc % 16 == 0, K % 32 == 0
// B 32x16 layout: lane holds column N=lane%16; lanes 0-15 K 0-7 / 16-23,
// lanes 16-31 K 8-15 / 24-31 (same order as A).
__global__ void pack_b_kernel(const float* __restrict__ src,
                              bf16_t* __restrict__ dst, int K, int Nc) {
  const int kblocks = K >> 5;
  long idx = (long)blockIdx.x * blockDim.x + threadIdx.x;
  if (idx >= (long)K * Nc) return;
  const int e    = (int)(idx & 15);
  const int lane = (int)((idx >> 4) & 31);
  const int kb   = (int)((idx >> 9) % kblocks);
  const int nt   = (int)(idx / ((long)kblocks << 9));
  const int col  = nt * 16 + (lane & 15);
  const int half = lane >> 4;
  const int krow = (kb << 5) + a_kk(e, half);
  dst[idx] = f2bf(src[(size_t)krow * Nc + col]);
}

// ---------------------------------------------------------------------------
// WMMA GEMM on packed operands: C[M,Nc] = A[M,K] @ B[K,Nc], f32 out.
// One wave computes a 16 x (16*NT) slab of C: A fetched once per K-step and
// reused across NT back-to-back v_wmma issues.
// ---------------------------------------------------------------------------
#define NT 4
__global__ void wmma_gemm_packed_kernel(const bf16_t* __restrict__ Ap,
                                        const bf16_t* __restrict__ Bp,
                                        float* __restrict__ C,
                                        int M, int K, int Nc) {
  const int wave = (int)((blockIdx.x * (long)blockDim.x + threadIdx.x) >> 5);
  const int lane = threadIdx.x & 31;
  const int kblocks = K >> 5;
  const int ngroups = Nc >> 6;               // groups of NT=4 tiles
  const int mtiles  = M >> 4;
  if (wave >= mtiles * ngroups) return;
  const int mt  = wave / ngroups;
  const int ng  = wave - mt * ngroups;

  const int half = lane >> 4;
  const int l16  = lane & 15;

  v8f acc0 = {}, acc1 = {}, acc2 = {}, acc3 = {};

  const bf16_t* ap = Ap + (((size_t)mt * kblocks) << 9) + (lane << 4);
  const bf16_t* bp = Bp + (((size_t)(ng * NT) * kblocks) << 9) + (lane << 4);
  const size_t bstride = (size_t)kblocks << 9;   // elements per B tile

  for (int kb = 0; kb < kblocks; ++kb) {
    const v16bf a  = *(const v16bf*)ap;
    const v16bf b0 = *(const v16bf*)(bp);
    const v16bf b1 = *(const v16bf*)(bp + bstride);
    const v16bf b2 = *(const v16bf*)(bp + 2 * bstride);
    const v16bf b3 = *(const v16bf*)(bp + 3 * bstride);
    acc0 = __builtin_amdgcn_wmma_f32_16x16x32_bf16(false, a, false, b0, (short)0, acc0, false, false);
    acc1 = __builtin_amdgcn_wmma_f32_16x16x32_bf16(false, a, false, b1, (short)0, acc1, false, false);
    acc2 = __builtin_amdgcn_wmma_f32_16x16x32_bf16(false, a, false, b2, (short)0, acc2, false, false);
    acc3 = __builtin_amdgcn_wmma_f32_16x16x32_bf16(false, a, false, b3, (short)0, acc3, false, false);
    ap += 512;
    bp += 512;
  }

  // C 16x16 layout: vgpr r -> row r (lanes 0-15) / r+8 (lanes 16-31), col l16
  float* cbase = C + (size_t)(mt * 16 + half * 8) * Nc + ng * (NT * 16) + l16;
#pragma unroll
  for (int r = 0; r < 8; ++r) {
    float* crow = cbase + (size_t)r * Nc;
    crow[0]  = acc0[r];
    crow[16] = acc1[r];
    crow[32] = acc2[r];
    crow[48] = acc3[r];
  }
}

// ---------------------------------------------------------------------------
// Per-node attention logits: as[n,h] = <h[n,h,:], a_src[h,:]>, same for ad
// ---------------------------------------------------------------------------
__global__ void node_alpha_kernel(const float* __restrict__ hfeat,
                                  const float* __restrict__ a_src,
                                  const float* __restrict__ a_dst,
                                  float* __restrict__ as_out,
                                  float* __restrict__ ad_out,
                                  int Nn, int H, int D) {
  int idx = blockIdx.x * blockDim.x + threadIdx.x;   // n*H + h
  if (idx >= Nn * H) return;
  const int n = idx / H, h = idx - n * H;
  const float* hp = hfeat + (size_t)n * H * D + (size_t)h * D;
  float ss = 0.0f, sd = 0.0f;
  for (int c = 0; c < D; ++c) {
    const float v = hp[c];
    ss += v * a_src[h * D + c];
    sd += v * a_dst[h * D + c];
  }
  as_out[idx] = ss;
  ad_out[idx] = sd;
}

// ---------------------------------------------------------------------------
// Edge passes. Edge id e < E reads edge_index (int64, [2,E]); e >= E is the
// implicit self-loop (s = d = e - E).
// ---------------------------------------------------------------------------
static __device__ __forceinline__ void edge_sd(const long long* ei, int E,
                                               long e, int* s, int* d) {
  if (e < E) { *s = (int)ei[e]; *d = (int)ei[(long)E + e]; }
  else       { *s = *d = (int)(e - E); }
}

__global__ void edge_max_kernel(const long long* __restrict__ ei, int E, long Etot,
                                const float* __restrict__ as,
                                const float* __restrict__ ad,
                                float* __restrict__ m, int H) {
  long e = (long)blockIdx.x * blockDim.x + threadIdx.x;
  if (e >= Etot) return;
  int s, d; edge_sd(ei, E, e, &s, &d);
  for (int h = 0; h < H; ++h) {
    float v = as[(long)s * H + h] + ad[(long)d * H + h];
    v = (v > 0.0f) ? v : NEG_SLOPE * v;
    atomicMaxFloat(&m[(long)d * H + h], v);
  }
}

__global__ void edge_denom_kernel(const long long* __restrict__ ei, int E, long Etot,
                                  const float* __restrict__ as,
                                  const float* __restrict__ ad,
                                  const float* __restrict__ m,
                                  float* __restrict__ denom, int H) {
  long e = (long)blockIdx.x * blockDim.x + threadIdx.x;
  if (e >= Etot) return;
  int s, d; edge_sd(ei, E, e, &s, &d);
  for (int h = 0; h < H; ++h) {
    float v = as[(long)s * H + h] + ad[(long)d * H + h];
    v = (v > 0.0f) ? v : NEG_SLOPE * v;
    atomicAdd(&denom[(long)d * H + h], __expf(v - m[(long)d * H + h]));
  }
}

// one block per edge; blockDim == H*D; thread c handles feature c
__global__ void edge_aggregate_kernel(const long long* __restrict__ ei, int E, long Etot,
                                      const float* __restrict__ as,
                                      const float* __restrict__ ad,
                                      const float* __restrict__ m,
                                      const float* __restrict__ denom,
                                      const float* __restrict__ hfeat,
                                      float* __restrict__ out, int H, int D) {
  long e = blockIdx.x;
  if (e >= Etot) return;
  int s, d; edge_sd(ei, E, e, &s, &d);
  const int c = threadIdx.x;
  const int h = c / D;
  float v = as[(long)s * H + h] + ad[(long)d * H + h];
  v = (v > 0.0f) ? v : NEG_SLOPE * v;
  const float alpha = __expf(v - m[(long)d * H + h]) /
                      (denom[(long)d * H + h] + 1e-16f);
  atomicAdd(&out[(size_t)d * H * D + c], hfeat[(size_t)s * H * D + c] * alpha);
}

// ---------------------------------------------------------------------------
// Host-side orchestration
// ---------------------------------------------------------------------------
static inline unsigned blocks_for(long n, int bs) {
  return (unsigned)((n + bs - 1) / bs);
}

extern "C" void kernel_launch(void* const* d_in, const int* in_sizes, int n_in,
                              void* d_out, int out_size, void* d_ws, size_t ws_size,
                              hipStream_t stream) {
  const float*     x        = (const float*)d_in[0];
  const long long* ei       = (const long long*)d_in[1];  // int64 per reference
  const float*     W1       = (const float*)d_in[2];
  const float*     a_src1   = (const float*)d_in[3];
  const float*     a_dst1   = (const float*)d_in[4];
  const float*     b1       = (const float*)d_in[5];
  const float*     W_mu     = (const float*)d_in[6];
  const float*     a_src_mu = (const float*)d_in[7];
  const float*     a_dst_mu = (const float*)d_in[8];
  const float*     b_mu     = (const float*)d_in[9];
  const float*     W_ls     = (const float*)d_in[10];
  const float*     a_src_ls = (const float*)d_in[11];
  const float*     a_dst_ls = (const float*)d_in[12];
  const float*     b_ls     = (const float*)d_in[13];

  const int IN = 128, F1 = 256, H1 = 4, D1 = 64, LAT = 64;
  const int Nn = in_sizes[0] / IN;          // 50000, multiple of 16
  const int E  = in_sizes[1] / 2;
  const long Etot = (long)E + Nn;

  // workspace carve-out
  char* base = (char*)d_ws;
  size_t off = 0;
  auto alloc = [&](size_t bytes) -> void* {
    off = (off + 255) & ~(size_t)255;
    void* p = base + off;
    off += bytes;
    return p;
  };

  bf16_t* Ap1   = (bf16_t*)alloc((size_t)Nn * IN * 2);   // x packed for WMMA
  bf16_t* Bp1   = (bf16_t*)alloc((size_t)IN * F1 * 2);   // W1 packed
  bf16_t* Bpm   = (bf16_t*)alloc((size_t)F1 * LAT * 2);  // W_mu packed
  bf16_t* Bpl   = (bf16_t*)alloc((size_t)F1 * LAT * 2);  // W_ls packed
  bf16_t* Ap2   = (bf16_t*)alloc((size_t)Nn * F1 * 2);   // relu(out1) packed
  float*  h1    = (float*) alloc((size_t)Nn * F1 * 4);
  float*  as1   = (float*) alloc((size_t)Nn * H1 * 4);
  float*  ad1   = (float*) alloc((size_t)Nn * H1 * 4);
  float*  m1    = (float*) alloc((size_t)Nn * H1 * 4);
  float*  den1  = (float*) alloc((size_t)Nn * H1 * 4);
  float*  acc1  = (float*) alloc((size_t)Nn * F1 * 4);
  float*  h2m   = (float*) alloc((size_t)Nn * LAT * 4);
  float*  h2l   = (float*) alloc((size_t)Nn * LAT * 4);
  float*  as2m  = (float*) alloc((size_t)Nn * 4);
  float*  ad2m  = (float*) alloc((size_t)Nn * 4);
  float*  m2m   = (float*) alloc((size_t)Nn * 4);
  float*  den2m = (float*) alloc((size_t)Nn * 4);
  float*  accm  = (float*) alloc((size_t)Nn * LAT * 4);
  float*  as2l  = (float*) alloc((size_t)Nn * 4);
  float*  ad2l  = (float*) alloc((size_t)Nn * 4);
  float*  m2l   = (float*) alloc((size_t)Nn * 4);
  float*  den2l = (float*) alloc((size_t)Nn * 4);
  float*  accl  = (float*) alloc((size_t)Nn * LAT * 4);

  const int BS = 256;
  const float NEG_INF = -__builtin_inff();

  // ---- pack operands into WMMA order (f32 -> bf16, one pass)
  pack_a_kernel<<<blocks_for((long)Nn * IN, BS), BS, 0, stream>>>(x, Ap1, Nn, IN);
  pack_b_kernel<<<blocks_for((long)IN * F1, BS), BS, 0, stream>>>(W1, Bp1, IN, F1);
  pack_b_kernel<<<blocks_for((long)F1 * LAT, BS), BS, 0, stream>>>(W_mu, Bpm, F1, LAT);
  pack_b_kernel<<<blocks_for((long)F1 * LAT, BS), BS, 0, stream>>>(W_ls, Bpl, F1, LAT);

  // ---- layer 1 GEMM: h1 = x @ W1  (WMMA bf16 -> f32)
  {
    long waves = (long)(Nn >> 4) * (F1 >> 6);
    wmma_gemm_packed_kernel<<<blocks_for(waves * 32, BS), BS, 0, stream>>>(
        Ap1, Bp1, h1, Nn, IN, F1);
  }

  // ---- layer 1 attention logits
  node_alpha_kernel<<<blocks_for((long)Nn * H1, BS), BS, 0, stream>>>(
      h1, a_src1, a_dst1, as1, ad1, Nn, H1, D1);

  // ---- layer 1 segment softmax + aggregate
  fill_f32_kernel<<<blocks_for((long)Nn * H1, BS), BS, 0, stream>>>(m1, NEG_INF, (long)Nn * H1);
  fill_f32_kernel<<<blocks_for((long)Nn * H1, BS), BS, 0, stream>>>(den1, 0.0f, (long)Nn * H1);
  fill_f32_kernel<<<blocks_for((long)Nn * F1, BS), BS, 0, stream>>>(acc1, 0.0f, (long)Nn * F1);
  edge_max_kernel<<<blocks_for(Etot, BS), BS, 0, stream>>>(ei, E, Etot, as1, ad1, m1, H1);
  edge_denom_kernel<<<blocks_for(Etot, BS), BS, 0, stream>>>(ei, E, Etot, as1, ad1, m1, den1, H1);
  edge_aggregate_kernel<<<(unsigned)Etot, H1 * D1, 0, stream>>>(
      ei, E, Etot, as1, ad1, m1, den1, h1, acc1, H1, D1);

  // ---- out1 = relu(acc1 + b1), packed straight into layer-2 A layout
  pack_a_relu_bias_kernel<<<blocks_for((long)Nn * F1, BS), BS, 0, stream>>>(
      acc1, b1, Ap2, Nn, F1);

  // ---- layer 2 GEMMs (mu, logstd)
  {
    long waves = (long)(Nn >> 4) * (LAT >> 6);
    wmma_gemm_packed_kernel<<<blocks_for(waves * 32, BS), BS, 0, stream>>>(
        Ap2, Bpm, h2m, Nn, F1, LAT);
    wmma_gemm_packed_kernel<<<blocks_for(waves * 32, BS), BS, 0, stream>>>(
        Ap2, Bpl, h2l, Nn, F1, LAT);
  }

  // ---- mu head (H=1, D=64)
  node_alpha_kernel<<<blocks_for((long)Nn, BS), BS, 0, stream>>>(
      h2m, a_src_mu, a_dst_mu, as2m, ad2m, Nn, 1, LAT);
  fill_f32_kernel<<<blocks_for((long)Nn, BS), BS, 0, stream>>>(m2m, NEG_INF, (long)Nn);
  fill_f32_kernel<<<blocks_for((long)Nn, BS), BS, 0, stream>>>(den2m, 0.0f, (long)Nn);
  fill_f32_kernel<<<blocks_for((long)Nn * LAT, BS), BS, 0, stream>>>(accm, 0.0f, (long)Nn * LAT);
  edge_max_kernel<<<blocks_for(Etot, BS), BS, 0, stream>>>(ei, E, Etot, as2m, ad2m, m2m, 1);
  edge_denom_kernel<<<blocks_for(Etot, BS), BS, 0, stream>>>(ei, E, Etot, as2m, ad2m, m2m, den2m, 1);
  edge_aggregate_kernel<<<(unsigned)Etot, LAT, 0, stream>>>(
      ei, E, Etot, as2m, ad2m, m2m, den2m, h2m, accm, 1, LAT);

  // ---- logstd head
  node_alpha_kernel<<<blocks_for((long)Nn, BS), BS, 0, stream>>>(
      h2l, a_src_ls, a_dst_ls, as2l, ad2l, Nn, 1, LAT);
  fill_f32_kernel<<<blocks_for((long)Nn, BS), BS, 0, stream>>>(m2l, NEG_INF, (long)Nn);
  fill_f32_kernel<<<blocks_for((long)Nn, BS), BS, 0, stream>>>(den2l, 0.0f, (long)Nn);
  fill_f32_kernel<<<blocks_for((long)Nn * LAT, BS), BS, 0, stream>>>(accl, 0.0f, (long)Nn * LAT);
  edge_max_kernel<<<blocks_for(Etot, BS), BS, 0, stream>>>(ei, E, Etot, as2l, ad2l, m2l, 1);
  edge_denom_kernel<<<blocks_for(Etot, BS), BS, 0, stream>>>(ei, E, Etot, as2l, ad2l, m2l, den2l, 1);
  edge_aggregate_kernel<<<(unsigned)Etot, LAT, 0, stream>>>(
      ei, E, Etot, as2l, ad2l, m2l, den2l, h2l, accl, 1, LAT);

  // ---- finalize: d_out = [mu | logstd]
  float* out = (float*)d_out;
  bias_out_kernel<<<blocks_for((long)Nn * LAT, BS), BS, 0, stream>>>(
      accm, b_mu, out, (long)Nn * LAT, LAT);
  bias_out_kernel<<<blocks_for((long)Nn * LAT, BS), BS, 0, stream>>>(
      accl, b_ls, out + (size_t)Nn * LAT, (long)Nn * LAT, LAT);
}
